// VarRNNBase_72189810311798
// MI455X (gfx1250) — matrix operand order repaired and checked by reading, not compiled, not run on hardware
//
#include <hip/hip_runtime.h>
#include <math.h>

// Problem sizes (fixed by the reference)
#define BB   32
#define TT   512
#define DD   512
#define HH   512
#define G4   2048            // 4*H
#define MR   (TT*BB)         // 16384 GEMM rows

typedef __attribute__((ext_vector_type(16))) _Float16 v16h;
typedef __attribute__((ext_vector_type(8)))  _Float16 v8h;
typedef __attribute__((ext_vector_type(8)))  float    v8f;
typedef __attribute__((ext_vector_type(4)))  unsigned u32x4;
typedef __attribute__((ext_vector_type(8)))  int      i32x8;
typedef __attribute__((ext_vector_type(4)))  int      i32x4;

#if defined(__has_builtin)
#if __has_builtin(__builtin_amdgcn_tensor_load_to_lds)
#define HAVE_TDM 1
#endif
#endif
#ifndef HAVE_TDM
#define HAVE_TDM 0
#endif

__device__ __forceinline__ float sigf(float x) { return 1.0f / (1.0f + __expf(-x)); }

// ---------------------------------------------------------------------------
// Prep kernels
// ---------------------------------------------------------------------------
__global__ void k_mask_x(const float* __restrict__ x, const float* __restrict__ mx,
                         _Float16* __restrict__ seq0) {
  size_t i = (size_t)blockIdx.x * 256 + threadIdx.x;   // T*B*D elements
  int k = (int)(i % DD);
  int b = (int)((i / DD) % BB);
  int t = (int)(i / ((size_t)DD * BB));
  seq0[i] = (_Float16)(x[((size_t)b * TT + t) * DD + k] * mx[b * DD + k]);
}

__global__ void k_cvt(const float* __restrict__ in, _Float16* __restrict__ out, int n) {
  int i = blockIdx.x * 256 + threadIdx.x;
  if (i < n) out[i] = (_Float16)in[i];
}

// ---------------------------------------------------------------------------
// gx = seq @ W_ih^T + bias     (WMMA GEMM, M=16384, N=2048 per dir, K=512/1024)
// A: [MR][K] f16 row-major; Bt: [2][G4][K] f16 (already N-major-by-K = B^T)
// ---------------------------------------------------------------------------
template <int K>
__global__ __launch_bounds__(128) void k_gemm_gx(const _Float16* __restrict__ A,
                                                 const _Float16* __restrict__ Bt,
                                                 const float* __restrict__ bias,
                                                 float* __restrict__ gx) {
  __shared__ __align__(32) _Float16 As[64 * 48];   // 32-K stage, rows padded to 48
  __shared__ __align__(32) _Float16 Bs[64 * 48];
  const int tid = threadIdx.x;
  const int lane = tid & 31, wave = tid >> 5;
  const int khalf = lane >> 4, l15 = lane & 15;
  const int mh = wave >> 1, nh = wave & 1;         // 4 waves -> 2x2 32x32 quadrants
  const int row0 = blockIdx.x * 64;
  const int col0 = blockIdx.y * 64;
  const int d = blockIdx.z;
  const _Float16* Ab = A + (size_t)row0 * K;
  const _Float16* Bb = Bt + ((size_t)d * G4 + col0) * K;

  v8f zero = {};
  v8f acc[2][2] = {{zero, zero}, {zero, zero}};

  for (int kk = 0; kk < K; kk += 32) {
    // Stage 64x32 of A and of Bt (each thread: two 16B chunks per matrix)
    for (int r = 0; r < 2; ++r) {
      int c = tid + r * 128;          // chunk id 0..255, 8 f16 each
      int rr = c >> 2;                // row 0..63
      int cc = (c & 3) * 8;           // k offset 0..31
      *(v8h*)&As[rr * 48 + cc] = *(const v8h*)&Ab[(size_t)rr * K + kk + cc];
      *(v8h*)&Bs[rr * 48 + cc] = *(const v8h*)&Bb[(size_t)rr * K + kk + cc];
    }
    __syncthreads();

    v16h af[2], bf[2];
    for (int mt = 0; mt < 2; ++mt) {
      int ar = mh * 32 + mt * 16 + l15;
      af[mt] = *(const v16h*)&As[ar * 48 + 16 * khalf];
    }
    for (int nt = 0; nt < 2; ++nt) {
      int br = nh * 32 + nt * 16 + l15;
      bf[nt] = *(const v16h*)&Bs[br * 48 + 16 * khalf];
    }
    for (int mt = 0; mt < 2; ++mt)
      for (int nt = 0; nt < 2; ++nt)
        acc[mt][nt] = __builtin_amdgcn_wmma_f32_16x16x32_f16(
            false, af[mt], false, bf[nt], (short)0, acc[mt][nt], false, false);
    __syncthreads();
  }

  for (int mt = 0; mt < 2; ++mt)
    for (int nt = 0; nt < 2; ++nt)
      for (int v = 0; v < 8; ++v) {
        int M = row0 + mh * 32 + mt * 16 + v + 8 * khalf;
        int N = col0 + nh * 32 + nt * 16 + l15;
        gx[((size_t)d * MR + M) * G4 + N] = acc[mt][nt][v] + bias[d * G4 + N];
      }
}

// ---------------------------------------------------------------------------
// Persistent recurrence kernel: 32 WGs (16 per direction), each owns 32 hidden
// columns; W_hh slice (128 rows x 512 K, f16) resident in LDS (TDM-loaded).
// ---------------------------------------------------------------------------
#define WS_H 528          // padded row stride (f16 elems): 512 + 16 (32B pad)

#if HAVE_TDM
__device__ __forceinline__ void tdm_load_rows(const _Float16* gsrc, unsigned lds_byte) {
  // 2D tile: 32 rows x 512 f16, contiguous rows (stride 512), with 32B LDS
  // padding after each 1024B row -> matches WS_H=528 layout.
  unsigned long long ga = (unsigned long long)(size_t)gsrc;
  u32x4 g0;
  g0.x = 1u;                                  // count=1, user descriptor
  g0.y = lds_byte;                            // LDS dest (byte offset)
  g0.z = (unsigned)ga;                        // global_addr[31:0]
  g0.w = (unsigned)(ga >> 32) | (2u << 30);   // global_addr[56:32] | type=2
  i32x8 g1;
  g1[0] = (int)((1u << 16)      // data_size = 2B
                | (1u << 20)    // pad_enable
                | (7u << 22)    // pad_interval: 256 dwords (one 1024B row)
                | (7u << 25));  // pad_amount:   8 dwords (32B)
  g1[1] = (int)(512u << 16);    // tensor_dim0 = 512
  g1[2] = (int)(32u << 16);     // tensor_dim1 = 32
  g1[3] = (int)(512u << 16);    // tile_dim0   = 512
  g1[4] = 32;                   // tile_dim1   = 32
  g1[5] = 512;                  // tensor_dim0_stride = 512
  g1[6] = 0;
  g1[7] = 0;
  i32x4 z4 = {0, 0, 0, 0};
#if __clang_major__ >= 23
  i32x8 z8 = {0, 0, 0, 0, 0, 0, 0, 0};
  __builtin_amdgcn_tensor_load_to_lds(g0, g1, z4, z4, z8, 0);
#else
  __builtin_amdgcn_tensor_load_to_lds(g0, g1, z4, z4, 0);
#endif
}
#endif

__global__ __launch_bounds__(128) void k_lstm(
    const float* __restrict__ gx,        // [2][MR][G4]  x-contribution + bias
    const _Float16* __restrict__ whh,    // [2][G4][HH] f16
    const float* __restrict__ mask_h,    // [BB][HH]  (this layer's row)
    const float* __restrict__ mask_out,  // [BB][2H]  (layer 0 only)
    _Float16* __restrict__ seq_next,     // [MR][2H] f16 (layer 0 output) or null
    float* __restrict__ outp,            // [BB][TT][2H] (layer 1 output) or null
    float* __restrict__ hn,              // [4][BB][HH]
    float* __restrict__ cn,              // [4][BB][HH]
    _Float16* __restrict__ hbuf,         // [2 phase][2 dir][BB][HH] f16 (masked h)
    unsigned* __restrict__ bar,          // [2] monotonic barrier counters
    int layer) {
  extern __shared__ __align__(32) char smem[];
  _Float16* whh_s = (_Float16*)smem;               // [128][WS_H]
  _Float16* h_s   = whh_s + 128 * WS_H;            // [32][WS_H]
  float*    gates = (float*)(h_s + 32 * WS_H);     // [32][128]
  float*    c_s   = gates + 32 * 128;              // [32][32]

  const int tid = threadIdx.x;
  const int lane = tid & 31, wave = tid >> 5;
  const int khalf = lane >> 4, l15 = lane & 15;
  const int wg = blockIdx.x;
  const int dir = wg >> 4;                         // 0=fwd, 1=bwd
  const int base = (wg & 15) * 32;                 // owned hidden columns

  // --- Load the 128-row W_hh slice (i,f,g,o x 32 cols) into LDS ---
#if HAVE_TDM
  if (wave == 0) {
    unsigned lbase = (unsigned)(size_t)whh_s;      // LDS aperture: low 32 bits
    for (int q = 0; q < 4; ++q) {
      const _Float16* src = whh + ((size_t)dir * G4 + q * 512 + base) * HH;
      tdm_load_rows(src, lbase + (unsigned)(q * 32 * WS_H * 2));
    }
    __builtin_amdgcn_s_wait_tensorcnt(0);
  }
#else
  for (int idx = tid; idx < 128 * 64; idx += 128) {  // 8192 chunks of 8 f16
    int lr = idx >> 6, kc = (idx & 63) * 8;
    int q = lr >> 5, r = lr & 31;
    *(v8h*)&whh_s[lr * WS_H + kc] =
        *(const v8h*)&whh[((size_t)dir * G4 + q * 512 + base + r) * HH + kc];
  }
#endif
  for (int idx = tid; idx < BB * 32; idx += 128) c_s[idx] = 0.0f;  // c0 = 0
  __syncthreads();

  v8f zero = {};
  for (int step = 0; step < TT; ++step) {
    const int t = (dir == 0) ? step : (TT - 1 - step);
    const int ph = step & 1;

    // (a) stage masked h (full [32][512]) from the exchange buffer
    for (int idx = tid; idx < (BB * HH) / 8; idx += 128) {
      int row = idx >> 6, kc = (idx & 63) * 8;
      *(v8h*)&h_s[row * WS_H + kc] =
          *(const v8h*)&hbuf[(((size_t)ph * 2 + dir) * BB + row) * HH + kc];
    }
    __syncthreads();

    // (b) gates_slice[32,128] = h_masked[32,512] @ whh_slice^T  (64 WMMA/wave)
    v8f acc[2][2] = {{zero, zero}, {zero, zero}};
    for (int kk = 0; kk < HH; kk += 32) {
      v16h af[2], bf[2];
      for (int mt = 0; mt < 2; ++mt) {
        int ar = mt * 16 + l15;
        af[mt] = *(const v16h*)&h_s[ar * WS_H + kk + 16 * khalf];
      }
      for (int nt = 0; nt < 2; ++nt) {
        int br = wave * 32 + nt * 16 + l15;
        bf[nt] = *(const v16h*)&whh_s[br * WS_H + kk + 16 * khalf];
      }
      for (int mt = 0; mt < 2; ++mt)
        for (int nt = 0; nt < 2; ++nt)
          acc[mt][nt] = __builtin_amdgcn_wmma_f32_16x16x32_f16(
              false, af[mt], false, bf[nt], (short)0, acc[mt][nt], false, false);
    }

    // (c) add x-contribution (gx) and park gates in LDS
    for (int mt = 0; mt < 2; ++mt)
      for (int nt = 0; nt < 2; ++nt)
        for (int v = 0; v < 8; ++v) {
          int M = mt * 16 + v + 8 * khalf;            // batch row
          int nl = wave * 32 + nt * 16 + l15;         // 0..127 slice column
          int gcol = (nl >> 5) * 512 + base + (nl & 31);
          gates[M * 128 + nl] =
              acc[mt][nt][v] + gx[((size_t)dir * MR + (size_t)t * BB + M) * G4 + gcol];
        }
    __syncthreads();

    // (d) LSTM cell for 32x32 owned cells; publish masked h, write outputs
    const int wph = (step + 1) & 1;
    for (int r = 0; r < 8; ++r) {
      int j = tid + r * 128;                 // 0..1023
      int b = j >> 5, cc = j & 31;
      float gi = sigf(gates[b * 128 + cc]);
      float gf = sigf(gates[b * 128 + 32 + cc]);
      float gg = tanhf(gates[b * 128 + 64 + cc]);
      float go = sigf(gates[b * 128 + 96 + cc]);
      float cnew = gf * c_s[b * 32 + cc] + gi * gg;
      float hnew = go * tanhf(cnew);
      c_s[b * 32 + cc] = cnew;
      int hcol = base + cc;
      hbuf[(((size_t)wph * 2 + dir) * BB + b) * HH + hcol] =
          (_Float16)(hnew * mask_h[b * HH + hcol]);
      int col = dir * HH + hcol;
      if (layer == 0) {
        seq_next[((size_t)t * BB + b) * (2 * HH) + col] =
            (_Float16)(hnew * mask_out[b * (2 * HH) + col]);
      } else {
        outp[((size_t)b * TT + t) * (2 * HH) + col] = hnew;
      }
      if (step == TT - 1) {
        size_t o = ((size_t)(layer * 2 + dir) * BB + b) * HH + hcol;
        hn[o] = hnew;
        cn[o] = cnew;
      }
    }
    __threadfence();
    __syncthreads();

    // (e) monotonic grid barrier across the 16 WGs of this direction
    if (tid == 0) {
      atomicAdd(&bar[dir], 1u);
      unsigned target = 16u * (unsigned)(step + 1);
      while (atomicAdd(&bar[dir], 0u) < target) __builtin_amdgcn_s_sleep(1);
      __threadfence();
    }
    __syncthreads();
  }
}

// ---------------------------------------------------------------------------
// Host-side orchestration (all on `stream`; graph-capture safe)
// ---------------------------------------------------------------------------
extern "C" void kernel_launch(void* const* d_in, const int* in_sizes, int n_in,
                              void* d_out, int out_size, void* d_ws, size_t ws_size,
                              hipStream_t stream) {
  (void)in_sizes; (void)n_in; (void)out_size; (void)ws_size;
  const float* x       = (const float*)d_in[0];
  const float* mask_x  = (const float*)d_in[1];
  const float* mask_o  = (const float*)d_in[2];
  const float* mask_h  = (const float*)d_in[3];
  const float* w_ih_l0 = (const float*)d_in[4];
  const float* w_hh_l0 = (const float*)d_in[5];
  const float* b_l0    = (const float*)d_in[6];
  const float* w_ih_l1 = (const float*)d_in[7];
  const float* w_hh_l1 = (const float*)d_in[8];
  const float* b_l1    = (const float*)d_in[9];

  char* ws = (char*)d_ws;
  // Workspace layout (bytes)
  _Float16* seq0 = (_Float16*)(ws + 0);                    //  16,777,216
  _Float16* seq1 = (_Float16*)(ws + 16777216);             //  33,554,432
  _Float16* wih0 = (_Float16*)(ws + 50331648);             //   4,194,304
  _Float16* whh0 = (_Float16*)(ws + 54525952);             //   4,194,304
  _Float16* wih1 = (_Float16*)(ws + 58720256);             //   8,388,608
  _Float16* whh1 = (_Float16*)(ws + 67108864);             //   4,194,304
  float*    gx   = (float*)   (ws + 71303168);             // 268,435,456
  _Float16* hbuf = (_Float16*)(ws + 339738624);            //     131,072
  unsigned* bar  = (unsigned*)(ws + 339869696);            //         256

  float* out = (float*)d_out;                              // [B][T][2H]
  float* hn  = out + (size_t)BB * TT * 2 * HH;             // [4][B][H]
  float* cn  = hn + (size_t)4 * BB * HH;

  const size_t lds_bytes = (size_t)(128 * WS_H + 32 * WS_H) * 2 +
                           (size_t)(32 * 128 + 32 * 32) * 4;  // 189,440 B

  // Prep: masked input + f16 weight copies
  k_mask_x<<<(TT * BB * DD) / 256, 256, 0, stream>>>(x, mask_x, seq0);
  k_cvt<<<(2 * G4 * DD) / 256, 256, 0, stream>>>(w_ih_l0, wih0, 2 * G4 * DD);
  k_cvt<<<(2 * G4 * HH) / 256, 256, 0, stream>>>(w_hh_l0, whh0, 2 * G4 * HH);
  k_cvt<<<(2 * G4 * 2 * HH) / 256, 256, 0, stream>>>(w_ih_l1, wih1, 2 * G4 * 2 * HH);
  k_cvt<<<(2 * G4 * HH) / 256, 256, 0, stream>>>(w_hh_l1, whh1, 2 * G4 * HH);

  // Layer 0
  k_gemm_gx<512><<<dim3(MR / 64, G4 / 64, 2), 128, 0, stream>>>(seq0, wih0, b_l0, gx);
  hipMemsetAsync(hbuf, 0, 131072 + 256, stream);           // h0=0 + barrier counters
  k_lstm<<<32, 128, lds_bytes, stream>>>(gx, whh0, mask_h, mask_o, seq1, nullptr,
                                         hn, cn, hbuf, bar, 0);
  // Layer 1
  k_gemm_gx<1024><<<dim3(MR / 64, G4 / 64, 2), 128, 0, stream>>>(seq1, wih1, b_l1, gx);
  hipMemsetAsync(hbuf, 0, 131072 + 256, stream);
  k_lstm<<<32, 128, lds_bytes, stream>>>(gx, whh1, mask_h + BB * HH, mask_o, nullptr,
                                         out, hn, cn, hbuf, bar, 1);
}
